// VQ_30442728194288
// MI455X (gfx1250) — matrix operand order repaired and checked by reading, not compiled, not run on hardware
//
#include <hip/hip_runtime.h>

// VQ-VAE forward for MI455X (gfx1250, wave32, WMMA bf16 -> f32).
// Pipeline: cvt->bf16 | GEMM1+relu | GEMM2+relu | GEMM3+bias | dist+argmin | gather+loss.
// B-operands (weights / codebook) are staged once per workgroup into padded LDS;
// the distance kernel runs M=32 per wave so each LDS B-fragment feeds 2 WMMAs.

#define NROWS  131072
#define IN_DIM 512
#define HID    64
#define ODIM   256
#define KPROTO 1024

typedef __attribute__((ext_vector_type(16))) __bf16 v16bf;
typedef __attribute__((ext_vector_type(8)))  __bf16 v8bf;
typedef __attribute__((ext_vector_type(8)))  float  v8f;

#define WMMA_BF16(a, b, c) \
  __builtin_amdgcn_wmma_f32_16x16x32_bf16(false, (a), false, (b), (short)0, (c), false, false)

__device__ __forceinline__ unsigned short f2bf(float f) {
  unsigned u = __float_as_uint(f);
  u += 0x7FFFu + ((u >> 16) & 1u);          // round-to-nearest-even
  return (unsigned short)(u >> 16);
}

__device__ __forceinline__ v16bf join8(v8bf lo, v8bf hi) {
  return __builtin_shufflevector(lo, hi, 0,1,2,3,4,5,6,7,8,9,10,11,12,13,14,15);
}

// A fragment, 16x32 bf16 (ISA 7.12.2): lane = m + 16*kh holds
// src[m, kb+kh*8 .. +7] in VGPRs 0-3 and src[m, kb+16+kh*8 .. +7] in VGPRs 4-7.
__device__ __forceinline__ v16bf load_a_frag(const unsigned short* src, int ld,
                                             int m0, int kb, int lane) {
  int m = lane & 15, kh = lane >> 4;
  const __bf16* p = (const __bf16*)(src + (size_t)(m0 + m) * ld + kb + kh * 8);
  return join8(*(const v8bf*)p, *(const v8bf*)(p + 16));
}

// B fragment, 32x16 bf16 with B[k][n] = W[n0+n][kb+k], W staged in LDS with
// padded row stride ldp (elements): lane = n + 16*kg reads k = kg*8..+7 and 16+kg*8..+7.
__device__ __forceinline__ v16bf load_b_frag_lds(const __bf16* w, int ldp,
                                                 int n0, int kb, int lane) {
  int n = lane & 15, kg = lane >> 4;
  const __bf16* p = w + (n0 + n) * ldp + kb + kg * 8;
  return join8(*(const v8bf*)p, *(const v8bf*)(p + 16));
}

// Cooperative copy of a [rows x cols] bf16 matrix (row-major, global) into LDS
// with padded row stride ldp (cols+8 keeps ds_load_b128 bank-conflict free).
__device__ __forceinline__ void stage_lds(__bf16* dst, int ldp,
                                          const unsigned short* src, size_t srow0,
                                          int rows, int cols) {
  int units_per_row = cols >> 3;                 // 16B units
  int total = rows * units_per_row;
  for (int u = threadIdx.x; u < total; u += blockDim.x) {
    int row = u / units_per_row, cu = u - row * units_per_row;
    const v8bf* g =
        (const v8bf*)((const __bf16*)src + (srow0 + row) * (size_t)cols + cu * 8);
    *(v8bf*)(dst + (size_t)row * ldp + cu * 8) = *g;
  }
}

// ---------------------------------------------------------------- conversions
__global__ void cvt_bf16_kernel(const float* __restrict__ src,
                                unsigned short* __restrict__ dst, int n) {
  for (int i = blockIdx.x * blockDim.x + threadIdx.x; i < n;
       i += gridDim.x * blockDim.x)
    dst[i] = f2bf(src[i]);
}

__global__ void pnorm_kernel(const float* __restrict__ protos,
                             float* __restrict__ pnorm) {
  int j = blockIdx.x * blockDim.x + threadIdx.x;
  if (j < KPROTO) {
    const float* p = protos + (size_t)j * ODIM;
    float s = 0.f;
    for (int c = 0; c < ODIM; ++c) { float v = p[c]; s += v * v; }
    pnorm[j] = s;
  }
}

// ------------------------------------------------- stage 1: h = relu(x W1^T + b1)
#define LDP1 (IN_DIM + 8)   // 520: row stride == 4 dwords mod 64 -> conflict-free
__global__ __launch_bounds__(256) void gemm1_kernel(
    const unsigned short* __restrict__ xb, const unsigned short* __restrict__ w1,
    const float* __restrict__ b1, unsigned short* __restrict__ h) {
  extern __shared__ char smem[];
  __bf16* sw = (__bf16*)smem;                       // 64 x LDP1
  stage_lds(sw, LDP1, w1, 0, HID, IN_DIM);
  __syncthreads();

  int lane = threadIdx.x & 31;
  int tile = blockIdx.x * 8 + (threadIdx.x >> 5);
  int m0 = tile * 16;
  v8f acc[4];
  #pragma unroll
  for (int t = 0; t < 4; ++t) acc[t] = (v8f)0.f;
  for (int kb = 0; kb < IN_DIM; kb += 32) {
    v16bf a = load_a_frag(xb, IN_DIM, m0, kb, lane);
    #pragma unroll
    for (int t = 0; t < 4; ++t) {
      v16bf b = load_b_frag_lds(sw, LDP1, t * 16, kb, lane);
      acc[t] = WMMA_BF16(a, b, acc[t]);
    }
  }
  int n = lane & 15, g = lane >> 4;
  #pragma unroll
  for (int t = 0; t < 4; ++t) {
    float bias = b1[t * 16 + n];
    #pragma unroll
    for (int r = 0; r < 8; ++r) {
      float v = acc[t][r] + bias;
      v = v > 0.f ? v : 0.f;
      h[(size_t)(m0 + r + 8 * g) * HID + t * 16 + n] = f2bf(v);
    }
  }
}

// ---------------------------------------------- stage 2: h2 = relu(h W2^T + b2)
#define LDP2 (HID + 8)      // 72
__global__ __launch_bounds__(256) void gemm2_kernel(
    const unsigned short* __restrict__ h, const unsigned short* __restrict__ w2,
    const float* __restrict__ b2, unsigned short* __restrict__ h2) {
  extern __shared__ char smem[];
  __bf16* sw = (__bf16*)smem;                       // 256 x LDP2
  stage_lds(sw, LDP2, w2, 0, ODIM, HID);
  __syncthreads();

  int lane = threadIdx.x & 31;
  int tile = blockIdx.x * 8 + (threadIdx.x >> 5);
  int m0 = tile * 16;
  v8f acc[16];
  #pragma unroll
  for (int t = 0; t < 16; ++t) acc[t] = (v8f)0.f;
  #pragma unroll
  for (int kb = 0; kb < HID; kb += 32) {
    v16bf a = load_a_frag(h, HID, m0, kb, lane);
    #pragma unroll
    for (int t = 0; t < 16; ++t) {
      v16bf b = load_b_frag_lds(sw, LDP2, t * 16, kb, lane);
      acc[t] = WMMA_BF16(a, b, acc[t]);
    }
  }
  int n = lane & 15, g = lane >> 4;
  #pragma unroll
  for (int t = 0; t < 16; ++t) {
    float bias = b2[t * 16 + n];
    #pragma unroll
    for (int r = 0; r < 8; ++r) {
      float v = acc[t][r] + bias;
      v = v > 0.f ? v : 0.f;
      h2[(size_t)(m0 + r + 8 * g) * ODIM + t * 16 + n] = f2bf(v);
    }
  }
}

// --------------------------------------------- stage 3: lat = h2 Wmu^T + bmu
#define LDP3 (ODIM + 8)     // 264
__global__ __launch_bounds__(256) void gemm3_kernel(
    const unsigned short* __restrict__ h2, const unsigned short* __restrict__ wmu,
    const float* __restrict__ bmu, float* __restrict__ latf,
    unsigned short* __restrict__ latb) {
  extern __shared__ char smem[];
  __bf16* sw = (__bf16*)smem;                       // 256 x LDP3 (132KB)
  stage_lds(sw, LDP3, wmu, 0, ODIM, ODIM);
  __syncthreads();

  int lane = threadIdx.x & 31;
  int tile = blockIdx.x * 8 + (threadIdx.x >> 5);
  int m0 = tile * 16;
  v8f acc[16];
  #pragma unroll
  for (int t = 0; t < 16; ++t) acc[t] = (v8f)0.f;
  for (int kb = 0; kb < ODIM; kb += 32) {
    v16bf a = load_a_frag(h2, ODIM, m0, kb, lane);
    #pragma unroll
    for (int t = 0; t < 16; ++t) {
      v16bf b = load_b_frag_lds(sw, LDP3, t * 16, kb, lane);
      acc[t] = WMMA_BF16(a, b, acc[t]);
    }
  }
  int n = lane & 15, g = lane >> 4;
  #pragma unroll
  for (int t = 0; t < 16; ++t) {
    float bias = bmu[t * 16 + n];
    #pragma unroll
    for (int r = 0; r < 8; ++r) {
      float v = acc[t][r] + bias;
      size_t o = (size_t)(m0 + r + 8 * g) * ODIM + t * 16 + n;
      latf[o] = v;
      latb[o] = f2bf(v);
    }
  }
}

// ------------- stage 4: argmin_j (||p_j||^2 - 2 lat . p_j)  (||lat||^2 const/row)
// Codebook staged in LDS in 8 chunks of 128 rows (66KB padded), shared by the
// block's 8 waves. Each wave owns M=32 rows: two resident A-tile sets (128
// VGPRs), so every LDS B-fragment feeds two back-to-back WMMAs.
#define PCHUNK_ROWS 128
__global__ __launch_bounds__(256) void vq_argmin_kernel(
    const unsigned short* __restrict__ latb, const unsigned short* __restrict__ pb,
    const float* __restrict__ pnorm, int* __restrict__ idxout) {
  extern __shared__ char smem[];
  __bf16* sp = (__bf16*)smem;                       // 128 x LDP3
  int lane = threadIdx.x & 31;
  int tile = blockIdx.x * 8 + (threadIdx.x >> 5);   // 32-row tiles
  int m0 = tile * 32;

  v16bf aF[2][8];
  #pragma unroll
  for (int hh = 0; hh < 2; ++hh)
    #pragma unroll
    for (int s = 0; s < 8; ++s)
      aF[hh][s] = load_a_frag(latb, ODIM, m0 + hh * 16, s * 32, lane);

  float mv[2][8];
  int mi[2][8];
  #pragma unroll
  for (int hh = 0; hh < 2; ++hh)
    #pragma unroll
    for (int r = 0; r < 8; ++r) { mv[hh][r] = 3.4e38f; mi[hh][r] = 0; }

  int n = lane & 15, g = lane >> 4;
  for (int ch = 0; ch < KPROTO / PCHUNK_ROWS; ++ch) {
    __syncthreads();                                 // buffer reuse guard
    stage_lds(sp, LDP3, pb, (size_t)ch * PCHUNK_ROWS, PCHUNK_ROWS, ODIM);
    __syncthreads();
    #pragma unroll
    for (int ptl = 0; ptl < PCHUNK_ROWS / 16; ++ptl) {
      v8f acc0 = (v8f)0.f, acc1 = (v8f)0.f;
      #pragma unroll
      for (int s = 0; s < 8; ++s) {
        v16bf b = load_b_frag_lds(sp, LDP3, ptl * 16, s * 32, lane);
        acc0 = WMMA_BF16(aF[0][s], b, acc0);
        acc1 = WMMA_BF16(aF[1][s], b, acc1);
      }
      int j = ch * PCHUNK_ROWS + ptl * 16 + n;
      float pn = pnorm[j];
      #pragma unroll
      for (int r = 0; r < 8; ++r) {
        float k0 = pn - 2.0f * acc0[r];
        if (k0 < mv[0][r]) { mv[0][r] = k0; mi[0][r] = j; }
        float k1 = pn - 2.0f * acc1[r];
        if (k1 < mv[1][r]) { mv[1][r] = k1; mi[1][r] = j; }
      }
    }
  }
  // Reduce over the 16 columns held by each 16-lane half (xor masks stay in-half).
  #pragma unroll
  for (int hh = 0; hh < 2; ++hh)
    #pragma unroll
    for (int r = 0; r < 8; ++r) {
      float v = mv[hh][r];
      int i = mi[hh][r];
      #pragma unroll
      for (int off = 1; off < 16; off <<= 1) {
        float ov = __shfl_xor(v, off, 32);
        int oi = __shfl_xor(i, off, 32);
        if (ov < v || (ov == v && oi < i)) { v = ov; i = oi; }
      }
      if (n == 0) idxout[m0 + hh * 16 + r + 8 * g] = i;
    }
}

// ------------------- stage 5: out = protos[idx]; partial sums of (q - lat)^2
__global__ __launch_bounds__(256) void gather_loss_kernel(
    const int* __restrict__ idx, const float* __restrict__ protos,
    const float* __restrict__ latf, float* __restrict__ out,
    float* __restrict__ partials) {
  __shared__ float sdata[256];
  const size_t total = (size_t)NROWS * ODIM;
  float s = 0.f;
  for (size_t i = (size_t)blockIdx.x * 256 + threadIdx.x; i < total;
       i += (size_t)gridDim.x * 256) {
    int row = (int)(i >> 8), col = (int)(i & 255);
    int j = idx[row];
    float qv = protos[(size_t)j * ODIM + col];
    out[i] = qv;
    float d = qv - latf[i];
    s += d * d;
  }
  sdata[threadIdx.x] = s;
  __syncthreads();
  for (int k = 128; k > 0; k >>= 1) {
    if (threadIdx.x < (unsigned)k) sdata[threadIdx.x] += sdata[threadIdx.x + k];
    __syncthreads();
  }
  if (threadIdx.x == 0) partials[blockIdx.x] = sdata[0];
}

__global__ __launch_bounds__(256) void finalize_kernel(
    const float* __restrict__ partials, int nb, float* __restrict__ tail) {
  __shared__ float sdata[256];
  float s = 0.f;
  for (int i = threadIdx.x; i < nb; i += 256) s += partials[i];
  sdata[threadIdx.x] = s;
  __syncthreads();
  for (int k = 128; k > 0; k >>= 1) {
    if (threadIdx.x < (unsigned)k) sdata[threadIdx.x] += sdata[threadIdx.x + k];
    __syncthreads();
  }
  if (threadIdx.x == 0) {
    float mse = sdata[0] / (float)((size_t)NROWS * ODIM);
    tail[0] = 1.25f * mse;  // beta*commitment + embedding, numerically equal terms
    tail[1] = 0.0f;         // capacity = 0
  }
}

// ------------------------------------------------------------------- launcher
extern "C" void kernel_launch(void* const* d_in, const int* in_sizes, int n_in,
                              void* d_out, int out_size, void* d_ws, size_t ws_size,
                              hipStream_t stream) {
  const float* x      = (const float*)d_in[0];
  const float* W1     = (const float*)d_in[1];
  const float* b1     = (const float*)d_in[2];
  const float* W2     = (const float*)d_in[3];
  const float* b2     = (const float*)d_in[4];
  const float* Wmu    = (const float*)d_in[5];
  const float* bmu    = (const float*)d_in[6];
  const float* protos = (const float*)d_in[7];
  float* out = (float*)d_out;

  char* w = (char*)d_ws;
  const size_t OFF_XB   = 0;  // x_bf16; later reused (dead) as f32 lat buffer
  const size_t OFF_W1   = OFF_XB  + (size_t)NROWS * IN_DIM * 2;
  const size_t OFF_W2   = OFF_W1  + (size_t)HID * IN_DIM * 2;
  const size_t OFF_WMU  = OFF_W2  + (size_t)ODIM * HID * 2;
  const size_t OFF_PB   = OFF_WMU + (size_t)ODIM * ODIM * 2;
  const size_t OFF_PN   = OFF_PB  + (size_t)KPROTO * ODIM * 2;
  const size_t OFF_H    = OFF_PN  + (size_t)KPROTO * 4;
  const size_t OFF_H2   = OFF_H   + (size_t)NROWS * HID * 2;
  const size_t OFF_LATB = OFF_H2  + (size_t)NROWS * ODIM * 2;
  const size_t OFF_IDX  = OFF_LATB + (size_t)NROWS * ODIM * 2;
  const size_t OFF_PART = OFF_IDX + (size_t)NROWS * 4;

  unsigned short* xb   = (unsigned short*)(w + OFF_XB);
  unsigned short* w1b  = (unsigned short*)(w + OFF_W1);
  unsigned short* w2b  = (unsigned short*)(w + OFF_W2);
  unsigned short* wmub = (unsigned short*)(w + OFF_WMU);
  unsigned short* pb   = (unsigned short*)(w + OFF_PB);
  float*          pn   = (float*)(w + OFF_PN);
  unsigned short* hbuf = (unsigned short*)(w + OFF_H);
  unsigned short* h2b  = (unsigned short*)(w + OFF_H2);
  unsigned short* latb = (unsigned short*)(w + OFF_LATB);
  int*            idxb = (int*)(w + OFF_IDX);
  float*          part = (float*)(w + OFF_PART);
  float*          latf = (float*)(w + OFF_XB);  // alias: x_bf16 dead after GEMM1

  cvt_bf16_kernel<<<4096, 256, 0, stream>>>(x,      xb,   NROWS * IN_DIM);
  cvt_bf16_kernel<<<64,   256, 0, stream>>>(W1,     w1b,  HID * IN_DIM);
  cvt_bf16_kernel<<<64,   256, 0, stream>>>(W2,     w2b,  ODIM * HID);
  cvt_bf16_kernel<<<64,   256, 0, stream>>>(Wmu,    wmub, ODIM * ODIM);
  cvt_bf16_kernel<<<256,  256, 0, stream>>>(protos, pb,   KPROTO * ODIM);
  pnorm_kernel<<<(KPROTO + 255) / 256, 256, 0, stream>>>(protos, pn);

  const int blocks16 = NROWS / 16 / 8;  // 8 waves (16-row tiles) per block
  gemm1_kernel<<<blocks16, 256, (size_t)HID * LDP1 * 2, stream>>>(xb, w1b, b1, hbuf);
  gemm2_kernel<<<blocks16, 256, (size_t)ODIM * LDP2 * 2, stream>>>(hbuf, w2b, b2, h2b);
  gemm3_kernel<<<blocks16, 256, (size_t)ODIM * LDP3 * 2, stream>>>(h2b, wmub, bmu,
                                                                   latf, latb);
  const int blocks32 = NROWS / 32 / 8;  // 8 waves (32-row tiles) per block
  vq_argmin_kernel<<<blocks32, 256, (size_t)PCHUNK_ROWS * LDP3 * 2, stream>>>(
      latb, pb, pn, idxb);

  gather_loss_kernel<<<2048, 256, 0, stream>>>(idxb, protos, latf, out, part);
  finalize_kernel<<<1, 256, 0, stream>>>(part, 2048, out + (size_t)NROWS * ODIM);
}
